// SelfAttention_850403525069
// MI455X (gfx1250) — matrix operand order repaired and checked by reading, not compiled, not run on hardware
//
#include <hip/hip_runtime.h>

// ---------------------------------------------------------------------------
// CDNA5 (gfx1250, wave32) self-attention pipeline, bf16 WMMA + fp32 softmax.
//   All GEMMs are "NT": A is MxK row-major, B is supplied transposed (NxK
//   row-major), so every WMMA fragment is two contiguous 16B loads per lane.
//   v3: two k-chunks in flight per loop iteration in DISJOINT register
//   buffers, so chunk-B global loads overlap chunk-A WMMAs (partial loadcnt
//   waits instead of s_wait_loadcnt 0 before every WMMA). Residual tile still
//   staged with async global->LDS copies hidden behind the whole K loop.
// ---------------------------------------------------------------------------

typedef __bf16 bf16;
typedef __attribute__((ext_vector_type(4)))  __bf16 v4bf;
typedef __attribute__((ext_vector_type(8)))  __bf16 v8bf;
typedef __attribute__((ext_vector_type(16))) __bf16 v16bf;
typedef __attribute__((ext_vector_type(8)))  float  v8f;

static __device__ __forceinline__ v8f zero8() {
  v8f z = {0.f, 0.f, 0.f, 0.f, 0.f, 0.f, 0.f, 0.f};
  return z;
}

// A fragment: 16x32 bf16 tile, rows m contiguous in K (row-major, stride lda).
// Lane layout (ISA 7.12.2): lane l -> m = l&15, k-halves at (l>>4)*8 and +16.
static __device__ __forceinline__ v16bf load_fragA(const bf16* tile, int lda, int lane) {
  const int m = lane & 15, half = lane >> 4;
  const bf16* p = tile + (long)m * lda + half * 8;
  union { v16bf v; v8bf h[2]; } u;
  u.h[0] = *(const v8bf*)(p);        // k = half*8 + [0..7]
  u.h[1] = *(const v8bf*)(p + 16);   // k = half*8 + 16 + [0..7]
  return u.v;
}

// B fragment: 32x16 bf16 tile given as B^T (NxK row-major, stride ldb).
// Lane l -> n = l&15, k-halves at (l>>4)*8 and +16 (same order as A).
static __device__ __forceinline__ v16bf load_fragB(const bf16* tileT, int ldb, int lane) {
  const int n = lane & 15, half = lane >> 4;
  const bf16* p = tileT + (long)n * ldb + half * 8;
  union { v16bf v; v8bf h[2]; } u;
  u.h[0] = *(const v8bf*)(p);
  u.h[1] = *(const v8bf*)(p + 16);
  return u.v;
}

static __device__ __forceinline__ v8f wmma_bf16(v16bf a, v16bf b, v8f c) {
  return __builtin_amdgcn_wmma_f32_16x16x32_bf16(
      /*neg_a=*/false, a, /*neg_b=*/false, b,
      /*c_mod=*/(short)0, c, /*reuse_a=*/false, /*reuse_b=*/false);
}

// ---------------------------------------------------------------------------
// fp32 -> bf16 conversion, vectorized 4-wide (b128 load, b64 store).
// ---------------------------------------------------------------------------
__global__ void cvt_f32_bf16_v4(const float* __restrict__ src, bf16* __restrict__ dst, long n4) {
  long i = (long)blockIdx.x * blockDim.x + threadIdx.x;
  const long stride = (long)gridDim.x * blockDim.x;
  for (; i < n4; i += stride) {
    const float4 f = ((const float4*)src)[i];
    v4bf o;
    o[0] = (bf16)f.x; o[1] = (bf16)f.y; o[2] = (bf16)f.z; o[3] = (bf16)f.w;
    ((v4bf*)dst)[i] = o;
  }
}

// ---------------------------------------------------------------------------
// Projection GEMM: D[m][n] = (bf16)( sum_k A[m][k]*Bt[n][k] + bias )
//   BIAS_ON_M=0: bias indexed by n (q/k proj), =1: bias indexed by m (v proj).
// Block = 128 threads (4 waves), block tile 128x64, wave tile 32x64.
// K-loop step 64: two 32-k chunks in disjoint buffers; chunk-B loads overlap
// chunk-A WMMAs.
// ---------------------------------------------------------------------------
template <int BIAS_ON_M>
__global__ void gemm_nt_bias(const bf16* __restrict__ A, const bf16* __restrict__ Bt,
                             const float* __restrict__ bias, bf16* __restrict__ D,
                             int K, int lda, int ldb, int ldd,
                             long strideA, long strideB, long strideD) {
  const int b = blockIdx.z;
  A  += (long)b * strideA;
  Bt += (long)b * strideB;
  D  += (long)b * strideD;
  const int lane = threadIdx.x & 31;
  const int wave = threadIdx.x >> 5;            // 0..3
  const int m0 = blockIdx.y * 128 + wave * 32;  // wave's 32-row strip
  const int n0 = blockIdx.x * 64;

  v8f acc[2][4];
#pragma unroll
  for (int i = 0; i < 2; ++i)
#pragma unroll
    for (int j = 0; j < 4; ++j) acc[i][j] = zero8();

  const bf16* Ar0 = A + (long)m0 * lda;
  const bf16* Ar1 = A + (long)(m0 + 16) * lda;
  for (int kk = 0; kk < K; kk += 64) {
    const int k2 = kk + 32;
    // Chunk A (k = kk..kk+31) and chunk B (k = kk+32..kk+63): issue all 24
    // b128 loads before any WMMA; disjoint buffers keep both sets live.
    v16bf a0A = load_fragA(Ar0 + kk, lda, lane);
    v16bf a1A = load_fragA(Ar1 + kk, lda, lane);
    v16bf bbA[4];
#pragma unroll
    for (int j = 0; j < 4; ++j)
      bbA[j] = load_fragB(Bt + (long)(n0 + j * 16) * ldb + kk, ldb, lane);
    v16bf a0B = load_fragA(Ar0 + k2, lda, lane);
    v16bf a1B = load_fragA(Ar1 + k2, lda, lane);
    v16bf bbB[4];
#pragma unroll
    for (int j = 0; j < 4; ++j)
      bbB[j] = load_fragB(Bt + (long)(n0 + j * 16) * ldb + k2, ldb, lane);
    // Chunk-A WMMAs need only a partial loadcnt wait (B loads still in flight).
#pragma unroll
    for (int j = 0; j < 4; ++j) {
      acc[0][j] = wmma_bf16(a0A, bbA[j], acc[0][j]);
      acc[1][j] = wmma_bf16(a1A, bbA[j], acc[1][j]);
    }
#pragma unroll
    for (int j = 0; j < 4; ++j) {
      acc[0][j] = wmma_bf16(a0B, bbB[j], acc[0][j]);
      acc[1][j] = wmma_bf16(a1B, bbB[j], acc[1][j]);
    }
  }

  const int n = lane & 15, half = lane >> 4;
#pragma unroll
  for (int i = 0; i < 2; ++i)
#pragma unroll
    for (int j = 0; j < 4; ++j)
#pragma unroll
      for (int r = 0; r < 8; ++r) {
        const int m  = m0 + i * 16 + r + half * 8;
        const int nn = n0 + j * 16 + n;
        const float bbv = BIAS_ON_M ? bias[m] : bias[nn];
        D[(long)m * ldd + nn] = (bf16)(acc[i][j][r] + bbv);
      }
}

// ---------------------------------------------------------------------------
// Fused scores + column-softmax -> beta^T (bf16).
//   S[s][t] = sum_o K[s][o] * Q[t][o]; softmax over t (== rows of S here).
// One block (256 thr, 8 waves) owns a 16-row s-strip; the full 16 x 2048 f32
// score strip lives in 128 KB LDS (CDNA5: 320 KB/WGP). Each loop iteration
// covers two tile-pairs in disjoint buffers: pair-B loads overlap pair-A
// WMMAs and LDS stores.
// ---------------------------------------------------------------------------
__global__ void scores_softmax(const bf16* __restrict__ Km, const bf16* __restrict__ Qm,
                               bf16* __restrict__ betaT, int Tn, int Ko) {
  extern __shared__ float Ss[];  // [16][Tn]
  const int b  = blockIdx.y;
  const int s0 = blockIdx.x * 16;
  const bf16* Kb = Km + (long)b * Tn * Ko;
  const bf16* Qb = Qm + (long)b * Tn * Ko;
  const int lane = threadIdx.x & 31;
  const int wave = threadIdx.x >> 5;  // 0..7

  // Preload the strip's A fragments once: rows s0..s0+15, o = 0..127.
  v16bf afr[4];
#pragma unroll
  for (int c = 0; c < 4; ++c)
    afr[c] = load_fragA(Kb + (long)s0 * Ko + c * 32, Ko, lane);

  const int n = lane & 15, half = lane >> 4;
  // Wave owns t-tiles {2w+16j, 2w+16j+1 : j=0..7}; two pairs per iteration.
  for (int j = 0; j < 8; j += 2) {
    const int tA = 2 * wave + 16 * j;
    const int tB = tA + 16;
    v16bf bA0[4], bA1[4], bB0[4], bB1[4];
#pragma unroll
    for (int c = 0; c < 4; ++c) {
      bA0[c] = load_fragB(Qb + (long)(tA * 16) * Ko + c * 32, Ko, lane);
      bA1[c] = load_fragB(Qb + (long)((tA + 1) * 16) * Ko + c * 32, Ko, lane);
    }
#pragma unroll
    for (int c = 0; c < 4; ++c) {
      bB0[c] = load_fragB(Qb + (long)(tB * 16) * Ko + c * 32, Ko, lane);
      bB1[c] = load_fragB(Qb + (long)((tB + 1) * 16) * Ko + c * 32, Ko, lane);
    }
    v8f acc0 = zero8(), acc1 = zero8();
#pragma unroll
    for (int c = 0; c < 4; ++c) {
      acc0 = wmma_bf16(afr[c], bA0[c], acc0);  // alternating accumulators
      acc1 = wmma_bf16(afr[c], bA1[c], acc1);
    }
#pragma unroll
    for (int r = 0; r < 8; ++r) {
      Ss[(r + half * 8) * Tn + tA * 16 + n]       = acc0[r];
      Ss[(r + half * 8) * Tn + (tA + 1) * 16 + n] = acc1[r];
    }
    v8f acc2 = zero8(), acc3 = zero8();
#pragma unroll
    for (int c = 0; c < 4; ++c) {
      acc2 = wmma_bf16(afr[c], bB0[c], acc2);
      acc3 = wmma_bf16(afr[c], bB1[c], acc3);
    }
#pragma unroll
    for (int r = 0; r < 8; ++r) {
      Ss[(r + half * 8) * Tn + tB * 16 + n]       = acc2[r];
      Ss[(r + half * 8) * Tn + (tB + 1) * 16 + n] = acc3[r];
    }
  }
  __syncthreads();

  // Row softmax: 16 threads per row, 16-lane shfl reductions.
  const int row = threadIdx.x >> 4;  // 0..15
  const int jj  = threadIdx.x & 15;
  float* Sr = Ss + row * Tn;
  float mx = -3.0e38f;
  for (int t = jj; t < Tn; t += 16) mx = fmaxf(mx, Sr[t]);
#pragma unroll
  for (int o = 8; o >= 1; o >>= 1) mx = fmaxf(mx, __shfl_xor(mx, o, 16));
  float sum = 0.f;
  for (int t = jj; t < Tn; t += 16) {
    float e = __expf(Sr[t] - mx);
    Sr[t] = e;
    sum += e;
  }
#pragma unroll
  for (int o = 8; o >= 1; o >>= 1) sum += __shfl_xor(sum, o, 16);
  const float inv = 1.0f / sum;

  bf16* outp = betaT + ((long)b * Tn + s0 + row) * Tn;
  for (int t = jj; t < Tn; t += 16) outp[t] = (bf16)(Sr[t] * inv);
}

// ---------------------------------------------------------------------------
// Output GEMM: out[b][c][s] = gamma * sum_t V[c][t]*betaT[s][t] + x[b][s][c]
// out is written in (B,C,T) flat layout == the reference's .view() result.
// Block tile 128(c) x 64(s), wave tile 32x64, two k-chunks in flight.
// The x residual tile (64s x 128c, 32 KB) is staged global->LDS with CDNA5
// async-to-LDS copies issued BEFORE the K loop; s_wait_asynccnt + barrier
// only at the epilogue, so the copy hides behind 2048 k-steps of WMMA work.
// ---------------------------------------------------------------------------
__global__ void out_gemm(const bf16* __restrict__ Vct, const bf16* __restrict__ betaT,
                         const float* __restrict__ x, const float* __restrict__ gamma,
                         float* __restrict__ out, int Tn, int C) {
  __shared__ float xs[64][132];  // rows 16B-aligned (132*4 = 528 = 33*16)
  const int b  = blockIdx.z;
  const int s0 = blockIdx.x * 64;
  const int c0 = blockIdx.y * 128;
  const bf16* A  = Vct + (long)b * C * Tn;     // C x T (K=t contiguous)
  const bf16* Bt = betaT + (long)b * Tn * Tn;  // T_s x T_t (K=t contiguous)
  const int lane = threadIdx.x & 31;
  const int wave = threadIdx.x >> 5;  // 0..3
  const int m0 = c0 + wave * 32;

  // Kick off async global->LDS staging of x[b][s0..+64][c0..+128].
  // 64 rows x 32 b128-chunks = 2048 chunks, 16 per thread.
  {
    const float* xb = x + ((long)b * Tn + s0) * C + c0;
    const unsigned lds_base = (unsigned)(unsigned long long)&xs[0][0];
    for (int chunk = threadIdx.x; chunk < 64 * 32; chunk += 128) {
      const int row = chunk >> 5, col4 = (chunk & 31) * 4;
      const float* gsrc = xb + (long)row * C + col4;
      const unsigned ldst = lds_base + (unsigned)(row * 132 + col4) * 4u;
      asm volatile("global_load_async_to_lds_b128 %0, %1, off"
                   :: "v"(ldst), "v"(gsrc) : "memory");
    }
  }

  v8f acc[2][4];
#pragma unroll
  for (int i = 0; i < 2; ++i)
#pragma unroll
    for (int j = 0; j < 4; ++j) acc[i][j] = zero8();

  const bf16* Ar0 = A + (long)m0 * Tn;
  const bf16* Ar1 = A + (long)(m0 + 16) * Tn;
  for (int kk = 0; kk < Tn; kk += 64) {
    const int k2 = kk + 32;
    v16bf a0A = load_fragA(Ar0 + kk, Tn, lane);
    v16bf a1A = load_fragA(Ar1 + kk, Tn, lane);
    v16bf bbA[4];
#pragma unroll
    for (int j = 0; j < 4; ++j)
      bbA[j] = load_fragB(Bt + (long)(s0 + j * 16) * Tn + kk, Tn, lane);
    v16bf a0B = load_fragA(Ar0 + k2, Tn, lane);
    v16bf a1B = load_fragA(Ar1 + k2, Tn, lane);
    v16bf bbB[4];
#pragma unroll
    for (int j = 0; j < 4; ++j)
      bbB[j] = load_fragB(Bt + (long)(s0 + j * 16) * Tn + k2, Tn, lane);
#pragma unroll
    for (int j = 0; j < 4; ++j) {
      acc[0][j] = wmma_bf16(a0A, bbA[j], acc[0][j]);
      acc[1][j] = wmma_bf16(a1A, bbA[j], acc[1][j]);
    }
#pragma unroll
    for (int j = 0; j < 4; ++j) {
      acc[0][j] = wmma_bf16(a0B, bbB[j], acc[0][j]);
      acc[1][j] = wmma_bf16(a1B, bbB[j], acc[1][j]);
    }
  }

  // Drain this wave's async copies, then make LDS visible block-wide.
  asm volatile("s_wait_asynccnt 0x0" ::: "memory");
  __syncthreads();

  const float g = gamma[0];
  const int n = lane & 15, half = lane >> 4;
  float* ob = out + (long)b * C * Tn;
#pragma unroll
  for (int i = 0; i < 2; ++i)
#pragma unroll
    for (int j = 0; j < 4; ++j)
#pragma unroll
      for (int r = 0; r < 8; ++r) {
        const int m  = m0 + i * 16 + r + half * 8;  // channel c
        const int nn = s0 + j * 16 + n;             // position s
        ob[(long)m * Tn + nn] = g * acc[i][j][r] + xs[nn - s0][m - c0];
      }
}

// ---------------------------------------------------------------------------
// Host launcher. Workspace (~146 MB): bf16 copies of x/Wq/Wk/Wv, Q, K,
// V (C,T), and betaT (B,T,T).
// ---------------------------------------------------------------------------
extern "C" void kernel_launch(void* const* d_in, const int* in_sizes, int n_in,
                              void* d_out, int out_size, void* d_ws, size_t ws_size,
                              hipStream_t stream) {
  (void)in_sizes; (void)n_in; (void)out_size; (void)ws_size;
  const int Bn = 8, Tn = 2048, Cc = 1024, C8 = 128;

  const float* x     = (const float*)d_in[0];
  const float* Wq    = (const float*)d_in[1];
  const float* bq    = (const float*)d_in[2];
  const float* Wk    = (const float*)d_in[3];
  const float* bk    = (const float*)d_in[4];
  const float* Wv    = (const float*)d_in[5];
  const float* bv    = (const float*)d_in[6];
  const float* gamma = (const float*)d_in[7];
  float* out = (float*)d_out;

  char* ws = (char*)d_ws;
  size_t off = 0;
  auto wsalloc = [&](size_t bytes) -> void* {
    void* p = (void*)(ws + off);
    off += (bytes + 255) & ~(size_t)255;
    return p;
  };

  const long nX    = (long)Bn * Tn * Cc;   // 16.8M
  const long nWqk  = (long)C8 * Cc;        // 131072
  const long nWv   = (long)Cc * Cc;        // 1.05M
  const long nQK   = (long)Bn * Tn * C8;   // 2.1M
  const long nBeta = (long)Bn * Tn * Tn;   // 33.6M

  bf16* xbf  = (bf16*)wsalloc(nX * 2);
  bf16* wqbf = (bf16*)wsalloc(nWqk * 2);
  bf16* wkbf = (bf16*)wsalloc(nWqk * 2);
  bf16* wvbf = (bf16*)wsalloc(nWv * 2);
  bf16* Qbf  = (bf16*)wsalloc(nQK * 2);
  bf16* Kbf  = (bf16*)wsalloc(nQK * 2);
  bf16* Vct  = (bf16*)wsalloc(nX * 2);     // (B, C, T)
  bf16* Beta = (bf16*)wsalloc(nBeta * 2);  // (B, T_s, T_t)

  // 1) fp32 -> bf16 conversions (4-wide vectorized).
  cvt_f32_bf16_v4<<<4096, 256, 0, stream>>>(x,  xbf,  nX / 4);
  cvt_f32_bf16_v4<<<128,  256, 0, stream>>>(Wq, wqbf, nWqk / 4);
  cvt_f32_bf16_v4<<<128,  256, 0, stream>>>(Wk, wkbf, nWqk / 4);
  cvt_f32_bf16_v4<<<1024, 256, 0, stream>>>(Wv, wvbf, nWv / 4);

  // 2) Projections.
  // Q[b][t][o]: A = x[b] (T x C), Bt = Wq (C8 x C), bias on n.
  dim3 gq(C8 / 64, Tn / 128, Bn);
  gemm_nt_bias<0><<<gq, 128, 0, stream>>>(xbf, wqbf, bq, Qbf,
      Cc, Cc, Cc, C8, (long)Tn * Cc, 0L, (long)Tn * C8);
  gemm_nt_bias<0><<<gq, 128, 0, stream>>>(xbf, wkbf, bk, Kbf,
      Cc, Cc, Cc, C8, (long)Tn * Cc, 0L, (long)Tn * C8);
  // V[b][c][t]: A = Wv (C x C), Bt = x[b] (T x C), bias on m -> D is (C, T).
  dim3 gv(Tn / 64, Cc / 128, Bn);
  gemm_nt_bias<1><<<gv, 128, 0, stream>>>(wvbf, xbf, bv, Vct,
      Cc, Cc, Cc, Tn, 0L, (long)Tn * Cc, (long)Cc * Tn);

  // 3) Fused scores + softmax(axis=t) -> betaT (bf16). 128 KB dynamic LDS.
  dim3 gs(Tn / 16, Bn);
  scores_softmax<<<gs, 256, 16 * Tn * sizeof(float), stream>>>(Kbf, Qbf, Beta, Tn, C8);

  // 4) out = gamma * V @ beta + x^T, emitted directly in (B,C,T) flat layout.
  dim3 go(Tn / 64, Cc / 128, Bn);
  out_gemm<<<go, 128, 0, stream>>>(Vct, Beta, x, gamma, out, Tn, Cc);
}
